// HybridRecurrentFFN_69123203662003
// MI455X (gfx1250) — matrix operand order repaired and checked by reading, not compiled, not run on hardware
//
#include <hip/hip_runtime.h>
#include <hip/hip_bf16.h>

// ---------------------------------------------------------------------------
// Problem constants (from the reference)
// ---------------------------------------------------------------------------
#define Hdim 1024
#define Rdim 2048
#define Ldim 2048
#define Bsz  4
#define Ssz  2048
#define Mtok (Bsz * Ssz)        // 8192 tokens

typedef __attribute__((ext_vector_type(16))) __bf16 v16bf;
typedef __attribute__((ext_vector_type(8)))  float  v8f;

// ---------------------------------------------------------------------------
// gfx1250 async global->LDS copy (probe-verified: builtin takes
// (int4-vec AS1*, int4-vec AS3*, imm offset, imm cpol))
// ---------------------------------------------------------------------------
#if defined(__has_builtin)
#if __has_builtin(__builtin_amdgcn_global_load_async_to_lds_b128)
#define USE_ASYNC_LDS 1
#endif
#endif
#ifndef USE_ASYNC_LDS
#define USE_ASYNC_LDS 0
#endif

typedef int v4i_gcc __attribute__((vector_size(16)));
typedef __attribute__((address_space(1))) v4i_gcc gbl_v4i;
typedef __attribute__((address_space(3))) v4i_gcc lds_v4i;

__device__ __forceinline__ void wait_async_lds() {
#if defined(__has_builtin)
#if __has_builtin(__builtin_amdgcn_s_wait_asynccnt)
    __builtin_amdgcn_s_wait_asynccnt(0);
#else
    asm volatile("s_wait_asynccnt 0x0" ::: "memory");
#endif
#endif
}

// ---------------------------------------------------------------------------
// Fast activation helpers (v_exp_f32 + v_rcp_f32; GEMMs are bf16 anyway)
// ---------------------------------------------------------------------------
__device__ __forceinline__ float rcp_(float x)     { return __builtin_amdgcn_rcpf(x); }
__device__ __forceinline__ float sigmoid_(float x) { return rcp_(1.0f + __expf(-x)); }
__device__ __forceinline__ float silu_(float x)    { return x * sigmoid_(x); }
__device__ __forceinline__ float tanh_(float x)    { return 2.0f * sigmoid_(2.0f * x) - 1.0f; }

__device__ __forceinline__ v16bf mk_frag(uint4 lo, uint4 hi) {
    union { uint4 u[2]; v16bf v; } t;
    t.u[0] = lo; t.u[1] = hi;
    return t.v;
}

// ---------------------------------------------------------------------------
// f32 -> bf16 conversion, 4 elements/thread (n must be a multiple of 4)
// ---------------------------------------------------------------------------
__global__ void cvt_f32_to_bf16_v4(const float4* __restrict__ in, uint2* __restrict__ out, int n4) {
    int i = blockIdx.x * blockDim.x + threadIdx.x;
    if (i < n4) {
        float4 v = in[i];
        union { __bf16 h[4]; uint2 u; } t;
        t.h[0] = (__bf16)v.x; t.h[1] = (__bf16)v.y;
        t.h[2] = (__bf16)v.z; t.h[3] = (__bf16)v.w;
        out[i] = t.u;
    }
}

// ---------------------------------------------------------------------------
// Tiled WMMA GEMM:  C[m,n] = sum_k A[m,k] * B[n,k]
//   A: [M,K] bf16 row-major (activations), B: [N,K] bf16 row-major (weights)
// Block: 256 threads = 8 waves. Block tile 128x128, K-step 64, double-buffered
// LDS (one barrier / K-step, 16 WMMAs / barrier). Wave (wm,wn) in 2x4 grid
// computes a 64x32 sub-tile = 4x2 WMMA 16x16 tiles.
//
// MODE 0: fused projection -> f32, activation by column group (n>>11):
//         0=tanh(f) 1=sigmoid(g) 2=silu(c) 3=identity(q)
// MODE 1: up/gate -> bf16, silu applied for n >= nhalf
// MODE 2: plain f32 store (rec branch -> d_out)
// MODE 3: f32 accumulate (local branch: d_out += v)
// ---------------------------------------------------------------------------
#define TILE_M 128
#define TILE_N 128
#define TILE_K 64
#define LDSS   72   // padded row stride in bf16 elements (144 B = 9x16B)

template <int MODE>
__global__ __launch_bounds__(256) void gemm_bf16_wmma(
    const __bf16* __restrict__ A, const __bf16* __restrict__ B,
    float* __restrict__ Cf, __bf16* __restrict__ Cb,
    int M, int N, int K, int nhalf)
{
    __shared__ alignas(16) __bf16 As[2][TILE_M * LDSS];
    __shared__ alignas(16) __bf16 Bs[2][TILE_N * LDSS];

    const int tid  = threadIdx.x;
    const int lane = tid & 31;
    const int wave = tid >> 5;
    const int wm   = wave & 1;        // 0..1
    const int wn   = wave >> 1;       // 0..3
    const int m0   = blockIdx.y * TILE_M;
    const int n0   = blockIdx.x * TILE_N;

    v8f acc[4][2];
    const v8f zero8 = {0.f, 0.f, 0.f, 0.f, 0.f, 0.f, 0.f, 0.f};
#pragma unroll
    for (int i = 0; i < 4; ++i)
#pragma unroll
        for (int j = 0; j < 2; ++j) acc[i][j] = zero8;

    // global->LDS mapping: thread t handles one row half: 32 bf16 = 4 x b128
    const int lrow = tid >> 1;          // 0..127
    const int lcol = (tid & 1) * 32;    // 0 or 32

    const int a_koff = (lane >> 4) * 8;   // A frag: K base 0 / 8  (and +16)
    const int b_koff = (lane >> 4) * 8;   // B frag: K base 0 / 8  (and +16)
    const int row16  = lane & 15;

    const int nk = K / TILE_K;

#if USE_ASYNC_LDS
    auto issue_tile = [&](int kt, int bsel) {
        const __bf16* gA = A + (size_t)(m0 + lrow) * K + kt * TILE_K + lcol;
        const __bf16* gB = B + (size_t)(n0 + lrow) * K + kt * TILE_K + lcol;
        __bf16* lA = &As[bsel][lrow * LDSS + lcol];
        __bf16* lB = &Bs[bsel][lrow * LDSS + lcol];
#pragma unroll
        for (int q = 0; q < 4; ++q) {
            __builtin_amdgcn_global_load_async_to_lds_b128(
                (gbl_v4i*)(gA + q * 8), (lds_v4i*)(lA + q * 8), 0, 0);
            __builtin_amdgcn_global_load_async_to_lds_b128(
                (gbl_v4i*)(gB + q * 8), (lds_v4i*)(lB + q * 8), 0, 0);
        }
    };
#else
    auto fetch_tile = [&](int kt, uint4* av, uint4* bv) {
        const uint4* gA = (const uint4*)(A + (size_t)(m0 + lrow) * K + kt * TILE_K + lcol);
        const uint4* gB = (const uint4*)(B + (size_t)(n0 + lrow) * K + kt * TILE_K + lcol);
#pragma unroll
        for (int q = 0; q < 4; ++q) { av[q] = gA[q]; bv[q] = gB[q]; }
    };
    auto store_tile = [&](int bsel, const uint4* av, const uint4* bv) {
        uint4* lA = (uint4*)&As[bsel][lrow * LDSS + lcol];
        uint4* lB = (uint4*)&Bs[bsel][lrow * LDSS + lcol];
#pragma unroll
        for (int q = 0; q < 4; ++q) { lA[q] = av[q]; lB[q] = bv[q]; }
    };
#endif

    auto compute = [&](int bsel) {
#pragma unroll
        for (int ks = 0; ks < TILE_K; ks += 32) {
            v16bf afrag[4], bfrag[2];
#pragma unroll
            for (int i = 0; i < 4; ++i) {
                const __bf16* pa = &As[bsel][(wm * 64 + i * 16 + row16) * LDSS + ks + a_koff];
                afrag[i] = mk_frag(*(const uint4*)pa, *(const uint4*)(pa + 16));
            }
#pragma unroll
            for (int j = 0; j < 2; ++j) {
                const __bf16* pb = &Bs[bsel][(wn * 32 + j * 16 + row16) * LDSS + ks + b_koff];
                bfrag[j] = mk_frag(*(const uint4*)pb, *(const uint4*)(pb + 16));
            }
#pragma unroll
            for (int i = 0; i < 4; ++i)
#pragma unroll
                for (int j = 0; j < 2; ++j)
                    acc[i][j] = __builtin_amdgcn_wmma_f32_16x16x32_bf16(
                        false, afrag[i], false, bfrag[j], (short)0, acc[i][j], false, false);
        }
    };

    // ---- double-buffered main loop ----
#if USE_ASYNC_LDS
    issue_tile(0, 0);
    wait_async_lds();
    __syncthreads();
    for (int kt = 0; kt < nk; ++kt) {
        const int cur = kt & 1;
        if (kt + 1 < nk) issue_tile(kt + 1, 1 - cur);
        compute(cur);
        wait_async_lds();
        __syncthreads();
    }
#else
    {
        uint4 av[4], bv[4];
        fetch_tile(0, av, bv);
        store_tile(0, av, bv);
    }
    __syncthreads();
    for (int kt = 0; kt < nk; ++kt) {
        const int cur = kt & 1;
        uint4 av[4], bv[4];
        if (kt + 1 < nk) fetch_tile(kt + 1, av, bv);
        compute(cur);
        if (kt + 1 < nk) store_tile(1 - cur, av, bv);
        __syncthreads();
    }
#endif

    // Epilogue: C/D layout — VGPR e: lanes 0-15 hold M=e, lanes 16-31 M=e+8; N = lane&15
    const int mbase = m0 + wm * 64;
    const int nbase = n0 + wn * 32;
    const int mAdd  = (lane & 16) ? 8 : 0;
#pragma unroll
    for (int i = 0; i < 4; ++i)
#pragma unroll
        for (int j = 0; j < 2; ++j)
#pragma unroll
            for (int e = 0; e < 8; ++e) {
                int m = mbase + i * 16 + mAdd + e;
                int n = nbase + j * 16 + row16;
                float v = acc[i][j][e];
                if (MODE == 0) {
                    int grp = n >> 11;
                    if (grp == 0)      v = tanh_(v);
                    else if (grp == 1) v = sigmoid_(v);
                    else if (grp == 2) v = silu_(v);
                    Cf[(size_t)m * N + n] = v;
                } else if (MODE == 1) {
                    if (n >= nhalf) v = silu_(v);
                    Cb[(size_t)m * N + n] = (__bf16)v;
                } else if (MODE == 2) {
                    Cf[(size_t)m * N + n] = v;
                } else { // MODE == 3
                    Cf[(size_t)m * N + n] += v;
                }
            }
}

// ---------------------------------------------------------------------------
// Diagonal recurrence + fused readout.
// fgcq: [Mtok, 4R] f32 with col groups [f|g|c|q] already activated.
// One thread per (b,r) channel; writes readout = bf16(silu(q*state)).
// ---------------------------------------------------------------------------
__global__ __launch_bounds__(256) void recurrence_kernel(
    const float* __restrict__ fgcq, const float* __restrict__ init_state,
    __bf16* __restrict__ readout)
{
    int ch = blockIdx.x * blockDim.x + threadIdx.x;   // 0 .. B*R-1
    int b = ch >> 11;           // / R
    int r = ch & (Rdim - 1);    // % R
    float state = init_state[r];
    const float* base = fgcq + (size_t)b * Ssz * (4 * Rdim) + r;
    __bf16* ro = readout + (size_t)b * Ssz * Rdim + r;
    for (int s = 0; s < Ssz; ++s) {
        const float* p = base + (size_t)s * (4 * Rdim);
        float f = p[0];
        float g = p[Rdim];
        float c = p[2 * Rdim];
        float q = p[3 * Rdim];
        state = f * state + g * c;
        ro[(size_t)s * Rdim] = (__bf16)silu_(q * state);
    }
}

// ---------------------------------------------------------------------------
// hidden = up * silu(gate)   (silu already applied in GEMM epilogue)
// upg: [Mtok, 2L] bf16 ; hidden: [Mtok, L] bf16
// ---------------------------------------------------------------------------
__global__ void swiglu_mul(const __bf16* __restrict__ upg, __bf16* __restrict__ hidden) {
    int idx = blockIdx.x * blockDim.x + threadIdx.x;  // over Mtok*L
    int m = idx >> 11;          // / L
    int r = idx & (Ldim - 1);
    float up = (float)upg[(size_t)m * (2 * Ldim) + r];
    float sg = (float)upg[(size_t)m * (2 * Ldim) + Ldim + r];
    hidden[idx] = (__bf16)(up * sg);
}

// ---------------------------------------------------------------------------
// Launch
// ---------------------------------------------------------------------------
extern "C" void kernel_launch(void* const* d_in, const int* in_sizes, int n_in,
                              void* d_out, int out_size, void* d_ws, size_t ws_size,
                              hipStream_t stream) {
    (void)in_sizes; (void)n_in; (void)out_size; (void)ws_size;

    const float* x      = (const float*)d_in[0];
    const float* Wf     = (const float*)d_in[1];
    const float* Wi     = (const float*)d_in[2];
    const float* Wv     = (const float*)d_in[3];
    const float* Wq     = (const float*)d_in[4];
    const float* Wrec   = (const float*)d_in[5];
    const float* Wup    = (const float*)d_in[6];
    const float* Wgate  = (const float*)d_in[7];
    const float* Wdown  = (const float*)d_in[8];
    const float* init_s = (const float*)d_in[9];
    float* out = (float*)d_out;

    // ---- workspace carving (256B aligned) ----
    char* ws = (char*)d_ws;
    size_t off = 0;
    auto carve = [&](size_t bytes) -> char* {
        char* p = ws + off;
        off = (off + bytes + 255) & ~(size_t)255;
        return p;
    };
    __bf16* x_bf    = (__bf16*)carve((size_t)Mtok * Hdim * 2);             // 16.8 MB
    __bf16* Wcat_bf = (__bf16*)carve((size_t)4 * Rdim * Hdim * 2);         // 16.8 MB
    __bf16* Wrec_bf = (__bf16*)carve((size_t)Hdim * Rdim * 2);             //  4.2 MB
    __bf16* Wupg_bf = (__bf16*)carve((size_t)2 * Ldim * Hdim * 2);         //  8.4 MB
    __bf16* Wdwn_bf = (__bf16*)carve((size_t)Hdim * Ldim * 2);             //  4.2 MB
    __bf16* readout = (__bf16*)carve((size_t)Mtok * Rdim * 2);             // 33.6 MB
    float*  fgcq    = (float*) carve((size_t)Mtok * 4 * Rdim * 4);         // 268  MB
    // upg + hidden alias the fgcq region (fgcq is dead after recurrence_kernel)
    __bf16* upg    = (__bf16*)fgcq;                                        // 67 MB
    __bf16* hidden = (__bf16*)((char*)fgcq + (size_t)Mtok * 2 * Ldim * 2); // 33.6 MB

    const int CT = 256;
    auto cvt = [&](const float* src, __bf16* dst, size_t n) {
        size_t n4 = n / 4;
        cvt_f32_to_bf16_v4<<<dim3((unsigned)((n4 + CT - 1) / CT)), dim3(CT), 0, stream>>>(
            (const float4*)src, (uint2*)dst, (int)n4);
    };

    // ---- 1. precision conversion (x + weight packing) ----
    const size_t WRH = (size_t)Rdim * Hdim;   // 2M elements per R x H weight
    cvt(x,     x_bf,              (size_t)Mtok * Hdim);
    cvt(Wf,    Wcat_bf + 0 * WRH, WRH);
    cvt(Wi,    Wcat_bf + 1 * WRH, WRH);
    cvt(Wv,    Wcat_bf + 2 * WRH, WRH);
    cvt(Wq,    Wcat_bf + 3 * WRH, WRH);
    cvt(Wrec,  Wrec_bf,           (size_t)Hdim * Rdim);
    cvt(Wup,   Wupg_bf + 0 * WRH, (size_t)Ldim * Hdim);
    cvt(Wgate, Wupg_bf + 1 * WRH, (size_t)Ldim * Hdim);
    cvt(Wdown, Wdwn_bf,           (size_t)Hdim * Ldim);

    // ---- 2. fused projection GEMM: -> fgcq (activated) ----
    gemm_bf16_wmma<0><<<dim3(4 * Rdim / TILE_N, Mtok / TILE_M), dim3(256), 0, stream>>>(
        x_bf, Wcat_bf, fgcq, nullptr, Mtok, 4 * Rdim, Hdim, 0);

    // ---- 3. recurrence + fused readout (silu(q*state)) -> bf16 ----
    recurrence_kernel<<<dim3(Bsz * Rdim / 256), dim3(256), 0, stream>>>(fgcq, init_s, readout);

    // ---- 4. up/gate GEMM (writes into fgcq region, safe after recurrence) ----
    gemm_bf16_wmma<1><<<dim3(2 * Ldim / TILE_N, Mtok / TILE_M), dim3(256), 0, stream>>>(
        x_bf, Wupg_bf, nullptr, upg, Mtok, 2 * Ldim, Hdim, Ldim);

    // ---- 5. hidden = up * silu(gate) ----
    swiglu_mul<<<dim3((unsigned)((size_t)Mtok * Ldim / 256)), dim3(256), 0, stream>>>(upg, hidden);

    // ---- 6. rec branch GEMM -> out (overwrite) ----
    gemm_bf16_wmma<2><<<dim3(Hdim / TILE_N, Mtok / TILE_M), dim3(256), 0, stream>>>(
        readout, Wrec_bf, out, nullptr, Mtok, Hdim, Rdim, 0);

    // ---- 7. local branch GEMM -> out (accumulate) ----
    gemm_bf16_wmma<3><<<dim3(Hdim / TILE_N, Mtok / TILE_M), dim3(256), 0, stream>>>(
        hidden, Wdwn_bf, out, nullptr, Mtok, Hdim, Ldim, 0);
}
